// GraphSage_58969900974116
// MI455X (gfx1250) — compile-verified, matrix operand order checked by reading, not run on hardware
//
#include <hip/hip_runtime.h>

typedef __attribute__((ext_vector_type(16))) __bf16 v16bf;
typedef __attribute__((ext_vector_type(8)))  float  v8f;
typedef __bf16 bf16;

union V16 { v16bf v; uint4 q[2]; };

constexpr int N_NODES  = 100000;
constexpr int N_EDGES  = 800000;
constexpr int NODE_DIM = 192;
constexpr int CONF_DIM = 64;

constexpr int MBLK  = 128;
constexpr int KSTEP = 32;
constexpr int NSCAN = (N_NODES + 255) / 256;   // 391 scan blocks

__device__ __forceinline__ float lrelu(float x) { return x >= 0.f ? x : 0.2f * x; }

// CDNA5 async global->LDS copy (ASYNCcnt-tracked DMA, bypasses VGPRs)
__device__ __forceinline__ void async_ld16(unsigned ldsOff, const void* gptr) {
    asm volatile("global_load_async_to_lds_b128 %0, %1, off"
                 :: "v"(ldsOff), "v"(gptr) : "memory");
}
__device__ __forceinline__ void wait_async0() {
    asm volatile("s_wait_asynccnt 0" ::: "memory");
}

// ---------- weight convert + transpose: src fp32 [K, Nn] -> dst bf16 [Nn, dstStride] at k offset
__global__ void cvt_w(const float* __restrict__ src, bf16* __restrict__ dst,
                      int K, int Nn, int dstStride, int kOff) {
    int idx = blockIdx.x * 256 + threadIdx.x;
    if (idx >= K * Nn) return;
    int k = idx / Nn, n = idx % Nn;
    dst[(size_t)n * dstStride + kOff + k] = (bf16)src[idx];
}

// ---------- X0 = concat(config, node_features) bf16 [N, 256]
__global__ void pack_x0(const float* __restrict__ cfg, const float* __restrict__ nodef,
                        bf16* __restrict__ X0) {
    int idx = blockIdx.x * 256 + threadIdx.x;
    int n = idx >> 8, c = idx & 255;
    float v = (c < CONF_DIM) ? cfg[(size_t)n * CONF_DIM + c]
                             : nodef[(size_t)n * NODE_DIM + (c - CONF_DIM)];
    X0[idx] = (bf16)v;
}

// ---------- Y = concat(config(from X0), xB) bf16 [N, 256]
__global__ void pack_y(const bf16* __restrict__ X0, const bf16* __restrict__ xB,
                       bf16* __restrict__ Y) {
    int idx = blockIdx.x * 256 + threadIdx.x;
    int n = idx >> 8, c = idx & 255;
    Y[idx] = (c < CONF_DIM) ? X0[(size_t)n * 256 + c]
                            : xB[(size_t)n * NODE_DIM + (c - CONF_DIM)];
}

// ================= CSR build (once; reused by both layers) =================
__global__ void zero2(int* __restrict__ a, int* __restrict__ b) {
    int i = blockIdx.x * 256 + threadIdx.x;
    if (i < N_NODES) { a[i] = 0; b[i] = 0; }
}

__global__ void edge_count(const int* __restrict__ edges, int* __restrict__ cnt) {
    int e = blockIdx.x * 256 + threadIdx.x;
    if (e < N_EDGES) atomicAdd(&cnt[edges[N_EDGES + e]], 1);
}

__global__ void scan1(const int* __restrict__ cnt, int* __restrict__ bsum) {
    __shared__ int s[256];
    int t = threadIdx.x, i = blockIdx.x * 256 + t;
    s[t] = (i < N_NODES) ? cnt[i] : 0;
    __syncthreads();
    for (int d = 128; d > 0; d >>= 1) {
        if (t < d) s[t] += s[t + d];
        __syncthreads();
    }
    if (t == 0) bsum[blockIdx.x] = s[0];
}

__global__ void scan2(int* __restrict__ bsum) {   // single block, 512 threads
    __shared__ int s[512];
    int t = threadIdx.x;
    int v = (t < NSCAN) ? bsum[t] : 0;
    s[t] = v;
    __syncthreads();
    for (int d = 1; d < 512; d <<= 1) {
        int x = (t >= d) ? s[t - d] : 0;
        __syncthreads();
        s[t] += x;
        __syncthreads();
    }
    if (t < NSCAN) bsum[t] = s[t] - v;            // exclusive block offsets
}

__global__ void scan3(const int* __restrict__ cnt, const int* __restrict__ bsum,
                      int* __restrict__ rowStart) {
    __shared__ int s[256];
    int t = threadIdx.x, i = blockIdx.x * 256 + t;
    int v = (i < N_NODES) ? cnt[i] : 0;
    s[t] = v;
    __syncthreads();
    for (int d = 1; d < 256; d <<= 1) {
        int x = (t >= d) ? s[t - d] : 0;
        __syncthreads();
        s[t] += x;
        __syncthreads();
    }
    if (i < N_NODES) rowStart[i] = bsum[blockIdx.x] + s[t] - v;
}

__global__ void edge_scatter_csr(const int* __restrict__ edges, const int* __restrict__ rowStart,
                                 int* __restrict__ cursor, int* __restrict__ nbr) {
    int e = blockIdx.x * 256 + threadIdx.x;
    if (e >= N_EDGES) return;
    int d = edges[N_EDGES + e];
    int pos = atomicAdd(&cursor[d], 1);
    nbr[rowStart[d] + pos] = edges[e];           // store src
}

// ---------- per-node max aggregation (atomic-free; order-independent => deterministic)
__global__ void seg_max(const int* __restrict__ rowStart, const int* __restrict__ cnt,
                        const int* __restrict__ nbr, const bf16* __restrict__ Y,
                        bf16* __restrict__ aggB) {
    int n = blockIdx.x, j = threadIdx.x;
    int s0 = rowStart[n], c = cnt[n];
    float m = -3.0e38f;
    if (c > 0) {
        int s = nbr[s0];
        for (int i = 0; i < c; ++i) {
            int snext = (i + 1 < c) ? nbr[s0 + i + 1] : s;
            __builtin_prefetch(&Y[(size_t)snext * 256], 0, 1);
            float v = (float)Y[(size_t)s * 256 + j];
            m = fmaxf(m, v);
            s = snext;
        }
    } else {
        m = 0.f;                                  // isolated node: -inf -> 0
    }
    aggB[(size_t)n * 256 + j] = (bf16)m;
}

// ================= WMMA GEMM, full-width blocks, async double-buffered LDS =================
// out = epilogue( [A0|A1] @ Wt^T + bias );  Nout == CT*16, grid.y == 1 (A streamed from HBM once).
template<int CT>
__global__ __launch_bounds__(256)
void gemm_bf16_wmma(const bf16* __restrict__ A0, int k0,
                    const bf16* __restrict__ A1, int k1,
                    const bf16* __restrict__ Wt,
                    const float* __restrict__ bias,
                    const float* __restrict__ resid,
                    float* __restrict__ outF, bf16* __restrict__ outB,
                    int M, int K, int actMode) {
    constexpr int Nout = CT * 16;
    __shared__ __align__(16) bf16 ldsA[2][MBLK * KSTEP];
    __shared__ __align__(16) bf16 ldsW[2][Nout * KSTEP];

    const int tid  = threadIdx.x;
    const int wave = tid >> 5;
    const int lane = tid & 31;
    const int half = (lane >> 4) & 1;
    const int mr   = lane & 15;
    const int row0 = blockIdx.x * MBLK;

    const unsigned aBase = (unsigned)(size_t)(&ldsA[0][0]);
    const unsigned wBase = (unsigned)(size_t)(&ldsW[0][0]);

    const v8f vzero = {0.f, 0.f, 0.f, 0.f, 0.f, 0.f, 0.f, 0.f};
    v8f acc[CT];
    #pragma unroll
    for (int c = 0; c < CT; ++c) acc[c] = vzero;

    // stage one K-slab (A: 128x32, W: Noutx32) via async DMA; 16B chunks
    auto stage = [&](int buf, int kb) {
        const bf16* Ap; int astr, kk;
        if (kb < k0) { Ap = A0; astr = k0; kk = kb; }
        else         { Ap = A1; astr = k1; kk = kb - k0; }
        #pragma unroll
        for (int i = 0; i < 2; ++i) {            // 512 chunks / 256 threads
            int chunk = tid + i * 256;
            int r = chunk >> 2, q = chunk & 3;
            int grow = row0 + r;
            if (grow >= M) grow = M - 1;         // clamp: rows >= M never stored
            async_ld16(aBase + (unsigned)(buf * MBLK * KSTEP + r * KSTEP + q * 8) * 2,
                       (const void*)(Ap + (size_t)grow * astr + kk + q * 8));
        }
        #pragma unroll
        for (int i = 0; i < CT / 4; ++i) {       // CT*64 chunks / 256 threads
            int chunk = tid + i * 256;
            int r = chunk >> 2, q = chunk & 3;
            async_ld16(wBase + (unsigned)(buf * Nout * KSTEP + r * KSTEP + q * 8) * 2,
                       (const void*)(Wt + (size_t)r * K + kb + q * 8));
        }
    };

    stage(0, 0);
    wait_async0();
    __syncthreads();

    const int nsteps = K / KSTEP;
    for (int ks = 0; ks < nsteps; ++ks) {
        const int cur = ks & 1;
        if (ks + 1 < nsteps) stage(cur ^ 1, (ks + 1) * KSTEP);  // overlap DMA with WMMA

        V16 ua;
        const bf16* pa = &ldsA[cur][(wave * 16 + mr) * KSTEP + half * 8];
        ua.q[0] = *(const uint4*)(pa);
        ua.q[1] = *(const uint4*)(pa + 16);

        // B fragments: 2-deep software pipeline so ds_load of tile c+1 overlaps WMMA of tile c
        V16 ub[2];
        {
            const bf16* pb = &ldsW[cur][(mr) * KSTEP + half * 8];
            ub[0].q[0] = *(const uint4*)(pb);
            ub[0].q[1] = *(const uint4*)(pb + 16);
        }
        #pragma unroll
        for (int c = 0; c < CT; ++c) {
            if (c + 1 < CT) {
                const bf16* pb = &ldsW[cur][((c + 1) * 16 + mr) * KSTEP + half * 8];
                ub[(c + 1) & 1].q[0] = *(const uint4*)(pb);
                ub[(c + 1) & 1].q[1] = *(const uint4*)(pb + 16);
            }
            acc[c] = __builtin_amdgcn_wmma_f32_16x16x32_bf16(
                false, ua.v, false, ub[c & 1].v, (short)0, acc[c], false, false);
        }

        if (ks + 1 < nsteps) {
            wait_async0();
            __syncthreads();
        }
    }

    // D layout: lane holds column n = c*16 + mr; rows m = half*8 + r
    const int mbase = row0 + wave * 16 + half * 8;
    #pragma unroll
    for (int c = 0; c < CT; ++c) {
        const int n = c * 16 + mr;
        const float bv = bias[n];
        #pragma unroll
        for (int r = 0; r < 8; ++r) {
            int m = mbase + r;
            if (m < M) {
                float v = acc[c][r] + bv;
                if (actMode) v = lrelu(v);
                if (resid) v += resid[(size_t)m * Nout + n];
                if (outF) outF[(size_t)m * Nout + n] = v;
                if (outB) outB[(size_t)m * Nout + n] = (bf16)v;
            }
        }
    }
}

extern "C" void kernel_launch(void* const* d_in, const int* in_sizes, int n_in,
                              void* d_out, int out_size, void* d_ws, size_t ws_size,
                              hipStream_t stream) {
    const float* nodef = (const float*)d_in[0];
    const float* cfg   = (const float*)d_in[1];
    const int*   edges = (const int*)d_in[2];
    const float* pw0   = (const float*)d_in[3];
    const float* pb0   = (const float*)d_in[4];
    const float* pw1   = (const float*)d_in[5];
    const float* pb1   = (const float*)d_in[6];

    char* ws = (char*)d_ws;
    size_t off = 0;
    auto alloc = [&](size_t bytes) -> void* {
        void* p = ws + off;
        off += (bytes + 255) & ~(size_t)255;
        return p;
    };

    bf16* W0t = (bf16*)alloc((size_t)192 * 256 * 2);
    bf16* W1t = (bf16*)alloc((size_t)192 * 192 * 2);
    bf16 *WSt[2], *M0t[2], *M1t[2];
    for (int l = 0; l < 2; ++l) {
        WSt[l] = (bf16*)alloc((size_t)256 * 512 * 2);   // [wl;wr]^T
        M0t[l] = (bf16*)alloc((size_t)192 * 256 * 2);
        M1t[l] = (bf16*)alloc((size_t)192 * 192 * 2);
    }
    bf16*  X0       = (bf16*) alloc((size_t)N_NODES * 256 * 2);
    bf16*  H        = (bf16*) alloc((size_t)N_NODES * 192 * 2);
    float* xF       = (float*)alloc((size_t)N_NODES * 192 * 4);
    bf16*  xB       = (bf16*) alloc((size_t)N_NODES * 192 * 2);
    bf16*  Yb       = (bf16*) alloc((size_t)N_NODES * 256 * 2);
    bf16*  aggB     = (bf16*) alloc((size_t)N_NODES * 256 * 2);
    bf16*  SB       = (bf16*) alloc((size_t)N_NODES * 256 * 2);
    int*   cnt      = (int*)  alloc((size_t)N_NODES * 4);
    int*   rowStart = (int*)  alloc((size_t)N_NODES * 4);
    int*   cursor   = (int*)  alloc((size_t)N_NODES * 4);
    int*   bsum     = (int*)  alloc((size_t)512 * 4);
    int*   nbr      = (int*)  alloc((size_t)N_EDGES * 4);

    dim3 blk(256);
    auto cvt = [&](const float* s, bf16* d, int K, int Nn, int str, int ko) {
        cvt_w<<<dim3((K * Nn + 255) / 256), blk, 0, stream>>>(s, d, K, Nn, str, ko);
    };

    cvt(pw0, W0t, 256, 192, 256, 0);
    cvt(pw1, W1t, 192, 192, 192, 0);
    const float *sbl[2], *mb0[2], *mb1[2];
    for (int l = 0; l < 2; ++l) {
        const float* wl = (const float*)d_in[7 + 7 * l + 0];
        sbl[l]          = (const float*)d_in[7 + 7 * l + 1];
        const float* wr = (const float*)d_in[7 + 7 * l + 2];
        const float* w0 = (const float*)d_in[7 + 7 * l + 3];
        mb0[l]          = (const float*)d_in[7 + 7 * l + 4];
        const float* w1 = (const float*)d_in[7 + 7 * l + 5];
        mb1[l]          = (const float*)d_in[7 + 7 * l + 6];
        cvt(wl, WSt[l], 256, 256, 512, 0);
        cvt(wr, WSt[l], 256, 256, 512, 256);
        cvt(w0, M0t[l], 256, 192, 256, 0);
        cvt(w1, M1t[l], 192, 192, 192, 0);
    }

    pack_x0<<<N_NODES, blk, 0, stream>>>(cfg, nodef, X0);

    // CSR build (once, reused by both layers)
    const int NB = (N_NODES + 255) / 256, EB = (N_EDGES + 255) / 256;
    zero2<<<NB, blk, 0, stream>>>(cnt, cursor);
    edge_count<<<EB, blk, 0, stream>>>(edges, cnt);
    scan1<<<NSCAN, blk, 0, stream>>>(cnt, bsum);
    scan2<<<1, dim3(512), 0, stream>>>(bsum);
    scan3<<<NSCAN, blk, 0, stream>>>(cnt, bsum, rowStart);
    edge_scatter_csr<<<EB, blk, 0, stream>>>(edges, rowStart, cursor, nbr);

    dim3 g1((N_NODES + MBLK - 1) / MBLK);

    // prenet: H = lrelu(X0 @ w0 + b0); x = lrelu(H @ w1 + b1)
    gemm_bf16_wmma<12><<<g1, blk, 0, stream>>>(X0, 256, nullptr, 0, W0t, pb0,
                                               nullptr, nullptr, H, N_NODES, 256, 1);
    gemm_bf16_wmma<12><<<g1, blk, 0, stream>>>(H, 192, nullptr, 0, W1t, pb1,
                                               nullptr, xF, xB, N_NODES, 192, 1);

    for (int l = 0; l < 2; ++l) {
        pack_y<<<N_NODES, blk, 0, stream>>>(X0, xB, Yb);
        seg_max<<<N_NODES, blk, 0, stream>>>(rowStart, cnt, nbr, Yb, aggB);

        // SAGE: SB = [agg | y] @ [wl; wr] + bl  (single K=512 GEMM)
        gemm_bf16_wmma<16><<<g1, blk, 0, stream>>>(aggB, 256, Yb, 256, WSt[l], sbl[l],
                                                   nullptr, nullptr, SB, N_NODES, 512, 0);
        // MLP: H = lrelu(SB @ mw0 + mb0)
        gemm_bf16_wmma<12><<<g1, blk, 0, stream>>>(SB, 256, nullptr, 0, M0t[l], mb0[l],
                                                   nullptr, nullptr, H, N_NODES, 256, 1);
        // x = x + lrelu(H @ mw1 + mb1); last layer writes d_out
        float* of = (l == 1) ? (float*)d_out : xF;
        gemm_bf16_wmma<12><<<g1, blk, 0, stream>>>(H, 192, nullptr, 0, M1t[l], mb1[l],
                                                   xF, of, xB, N_NODES, 192, 1);
    }
}
